// VanillaLSTM_65481071411024
// MI455X (gfx1250) — matrix operand order repaired and checked by reading, not compiled, not run on hardware
//
#include <hip/hip_runtime.h>
#include <hip/hip_bf16.h>
#include <stdint.h>

typedef __attribute__((ext_vector_type(16))) __bf16 v16bf;
typedef __attribute__((ext_vector_type(8)))  float  v8f;
typedef __attribute__((ext_vector_type(4)))  int    v4i;

#define HID    256
#define NB     16
#define TSTEPS 255   // S-1
#define VOCAB  32000
#define MCHUNK 51    // M-tiles per blockIdx.y (5 * 51 = 255)

// ---------------------------------------------------------------------------
// CDNA5 async global->LDS copy (ASYNCcnt) with inline-asm fallback.
// ROCm signature (from diagnostics): (v4i AS1* src, v4i AS3* dst, Ii, Ii)
// ---------------------------------------------------------------------------
typedef __attribute__((address_space(1))) v4i glob_v4i;
typedef __attribute__((address_space(3))) v4i lds_v4i;

__device__ __forceinline__ void async_copy_b128(const void* gsrc, void* lds_dst) {
#if __has_builtin(__builtin_amdgcn_global_load_async_to_lds_b128)
  __builtin_amdgcn_global_load_async_to_lds_b128(
      (glob_v4i*)(uintptr_t)gsrc,
      (lds_v4i*)(uint32_t)(uintptr_t)lds_dst, 0, 0);
#else
  uint32_t lds_off = (uint32_t)(uintptr_t)lds_dst;
  asm volatile("global_load_async_to_lds_b128 %0, %1, off"
               :: "v"(lds_off), "v"(gsrc) : "memory");
#endif
}

__device__ __forceinline__ void wait_async0() {
#if __has_builtin(__builtin_amdgcn_s_wait_asynccnt)
  __builtin_amdgcn_s_wait_asynccnt(0);
#else
  asm volatile("s_wait_asynccnt 0x0" ::: "memory");
#endif
}

// ---------------------------------------------------------------------------
// Pack an fp32 weight matrix W[N][256] (used as B[k][n] = W[n][k]) into bf16
// WMMA B-fragment layout (ISA 7.12.2, 16-bit B 32x16):
//   Wp[((ntile*8 + kt)*32 + lane)*16 + e]
//   n = ntile*16 + (lane&15),  k = kt*32 + (e&7) + 8*(lane>>4) + 16*(e>>3)
// ---------------------------------------------------------------------------
__global__ void pack_b_bf16(const float* __restrict__ W, __bf16* __restrict__ Wp,
                            int total) {
  int idx = blockIdx.x * blockDim.x + threadIdx.x;
  if (idx >= total) return;
  int e     = idx & 15;
  int lane  = (idx >> 4) & 31;
  int kt    = (idx >> 9) & 7;
  int ntile = idx >> 12;
  int n = ntile * 16 + (lane & 15);
  int k = kt * 32 + (e & 7) + 8 * (lane >> 4) + ((e >> 3) << 4);
  Wp[idx] = (__bf16)W[(long)n * HID + k];
}

// ---------------------------------------------------------------------------
// Gx[(t*16+b)*1024 + j] = emb_eff[old[b][t]] . W_ih[j,:] + b_ih[j] + b_hh[j]
// One wave per (mtile=t, ntile) 16x16 tile; K loop 8x32 bf16 WMMA.
// A-fragment layout (ISA 7.12.2, 16-bit A 16x32):
//   row = lane&15, K(e) = kt*32 + e + 8*(lane>>4) + 8*(e>>3)
// ---------------------------------------------------------------------------
__global__ __launch_bounds__(256)
void gx_kernel(const int* __restrict__ old_tok, const float* __restrict__ emb,
               const __bf16* __restrict__ WihP,
               const float* __restrict__ b_ih, const float* __restrict__ b_hh,
               float* __restrict__ Gx) {
  int wave  = (blockIdx.x * blockDim.x + threadIdx.x) >> 5;
  int lane  = threadIdx.x & 31;
  int mtile = wave >> 6;          // t
  int ntile = wave & 63;
  if (mtile >= TSTEPS) return;
  int row = lane & 15, hi = lane >> 4;

  int tok = old_tok[row * 256 + mtile];          // old[b][t], b = row
  const float* erow = emb + (long)tok * HID;

  v8f acc = {};
  for (int kt = 0; kt < 8; ++kt) {
    v16bf a;
#pragma unroll
    for (int e = 0; e < 16; ++e) {
      int K = kt * 32 + e + 8 * hi + ((e >> 3) << 3);
      float v = erow[K];
      if (tok == 0) v = 0.0f;                    // emb.at[0].set(0)
      a[e] = (__bf16)v;
    }
    v16bf b = *(const v16bf*)(WihP + (((long)(ntile * 8 + kt) * 32 + lane) << 4));
    acc = __builtin_amdgcn_wmma_f32_16x16x32_bf16(false, a, false, b,
                                                  (short)0, acc, false, false);
  }
  int col = ntile * 16 + row;
  float bias = b_ih[col] + b_hh[col];
#pragma unroll
  for (int r = 0; r < 8; ++r) {
    int m = mtile * 16 + r + 8 * hi;             // m = t*16 + b
    Gx[(long)m * 1024 + col] = acc[r] + bias;
  }
}

// ---------------------------------------------------------------------------
// Sequential LSTM recurrence. Single 1024-thread workgroup (32 waves).
// LDS: h (16KB) + c (16KB) + gates (64KB) = 96KB of the 320KB WGP budget.
// ---------------------------------------------------------------------------
__global__ __launch_bounds__(1024)
void lstm_rec_kernel(const float* __restrict__ Gx, const __bf16* __restrict__ WhhP,
                     __bf16* __restrict__ hsP) {
  __shared__ float sh[NB * HID];
  __shared__ float sc[NB * HID];
  __shared__ float sg[NB * 1024];
  int tid = threadIdx.x;
  for (int i = tid; i < NB * HID; i += 1024) { sh[i] = 0.0f; sc[i] = 0.0f; }
  __syncthreads();

  int lane = tid & 31, wave = tid >> 5;
  int row = lane & 15, hi = lane >> 4;

  for (int t = 0; t < TSTEPS; ++t) {
#pragma unroll
    for (int nt2 = 0; nt2 < 2; ++nt2) {
      int ntile = wave * 2 + nt2;
      v8f acc = {};
      for (int kt = 0; kt < 8; ++kt) {
        v16bf a;
#pragma unroll
        for (int e = 0; e < 16; ++e) {
          int K = kt * 32 + e + 8 * hi + ((e >> 3) << 3);
          a[e] = (__bf16)sh[row * HID + K];
        }
        v16bf b = *(const v16bf*)(WhhP + (((long)(ntile * 8 + kt) * 32 + lane) << 4));
        acc = __builtin_amdgcn_wmma_f32_16x16x32_bf16(false, a, false, b,
                                                      (short)0, acc, false, false);
      }
      int col = ntile * 16 + row;
#pragma unroll
      for (int r = 0; r < 8; ++r) {
        int bb = r + 8 * hi;
        sg[bb * 1024 + col] = acc[r] + Gx[(long)(t * 16 + bb) * 1024 + col];
      }
    }
    __syncthreads();

#pragma unroll
    for (int i = 0; i < 4; ++i) {
      int idx = tid + i * 1024;                  // 16*256 = 4096 elements
      int bb = idx >> 8, k = idx & 255;
      float gi = sg[bb * 1024 + k];
      float gf = sg[bb * 1024 + 256 + k];
      float gg = sg[bb * 1024 + 512 + k];
      float go = sg[bb * 1024 + 768 + k];
      float si = 1.0f / (1.0f + __expf(-gi));
      float sf = 1.0f / (1.0f + __expf(-gf));
      float so = 1.0f / (1.0f + __expf(-go));
      float c  = sf * sc[idx] + si * tanhf(gg);
      sc[idx] = c;
      float h = so * tanhf(c);
      sh[idx] = h;
      // scatter h into bf16 A-fragment layout: invert K(e,hi) mapping
      int kt   = k >> 5, kk = k & 31;
      int hi2  = (kk >> 3) & 1;
      int e    = (kk & 7) + ((kk >> 4) << 3);
      int lane2 = (hi2 << 4) | bb;
      hsP[(((long)(t * 8 + kt) * 32 + lane2) << 4) + e] = (__bf16)h;
    }
    __syncthreads();
  }
}

// ---------------------------------------------------------------------------
// logits = hs @ W_lin^T + b_lin.
// Per block: 8 waves, each owning 4 N-tiles. The shared A stream (hs) for a
// PAIR of M-tiles is staged into LDS once per block with async global->LDS
// copies (double-buffered, overlapped with WMMA). Each B fragment feeds two
// WMMAs (M-pairing), halving W_lin L2 traffic.
// ---------------------------------------------------------------------------
__global__ __launch_bounds__(256)
void logits_kernel(const __bf16* __restrict__ hsP, const __bf16* __restrict__ WlinP,
                   const float* __restrict__ b_lin, float* __restrict__ out) {
  __shared__ __bf16 sA[2][2 * 8 * 32 * 16];    // 2 buffers x (2 M-tiles = 16KB)
  int tid  = threadIdx.x;
  int lane = tid & 31;
  int wave = tid >> 5;
  int m0 = blockIdx.y * MCHUNK;
  int m1 = m0 + MCHUNK;
  int nquad = blockIdx.x * 8 + wave;
  bool active = nquad < (VOCAB / 16) / 4;      // 500 quads
  if (!active) nquad = (VOCAB / 16) / 4 - 1;   // clamp; stay in barriers
  int ntbase = nquad * 4;
  int row = lane & 15, hi = lane >> 4;

  float bias[4];
#pragma unroll
  for (int q = 0; q < 4; ++q) bias[q] = b_lin[(ntbase + q) * 16 + row];

  // cooperative stage of one M-pair (8192 bf16 = 16KB) into LDS buffer `buf`
  auto stage = [&](int buf, int mt) {
    const __bf16* src = hsP + (long)mt * 4096;
#pragma unroll
    for (int j = 0; j < 4; ++j) {
      int chunk = tid + j * 256;               // 1024 chunks x 16B
      async_copy_b128(src + chunk * 8, &sA[buf][chunk * 8]);
    }
  };

  const int npairs = (MCHUNK + 1) / 2;         // 26 (odd tail handled by guard)
  stage(0, m0);
  for (int p = 0; p < npairs; ++p) {
    int mt = m0 + 2 * p;
    wait_async0();
    __syncthreads();
    if (p + 1 < npairs) stage((p + 1) & 1, mt + 2);

    const __bf16* Ab = sA[p & 1];
    v8f acc0[4] = {v8f{}, v8f{}, v8f{}, v8f{}};
    v8f acc1[4] = {v8f{}, v8f{}, v8f{}, v8f{}};
    for (int kt = 0; kt < 8; ++kt) {
      v16bf a0 = *(const v16bf*)(Ab + ((kt * 32 + lane) << 4));
      v16bf a1 = *(const v16bf*)(Ab + (((8 + kt) * 32 + lane) << 4));
#pragma unroll
      for (int q = 0; q < 4; ++q) {
        v16bf b = *(const v16bf*)(WlinP +
                   ((((long)(ntbase + q) * 8 + kt) * 32 + lane) << 4));
        acc0[q] = __builtin_amdgcn_wmma_f32_16x16x32_bf16(false, a0, false, b,
                                                          (short)0, acc0[q], false, false);
        acc1[q] = __builtin_amdgcn_wmma_f32_16x16x32_bf16(false, a1, false, b,
                                                          (short)0, acc1[q], false, false);
      }
    }
    if (active) {
#pragma unroll
      for (int q = 0; q < 4; ++q) {
        int v = (ntbase + q) * 16 + row;
#pragma unroll
        for (int r = 0; r < 8; ++r) {
          int batch = r + 8 * hi;
          out[((long)batch * TSTEPS + mt) * VOCAB + v] = acc0[q][r] + bias[q];
          if (mt + 1 < m1)
            out[((long)batch * TSTEPS + (mt + 1)) * VOCAB + v] = acc1[q][r] + bias[q];
        }
      }
    }
    __syncthreads();   // all readers done before this buffer is restaged
  }
}

// ---------------------------------------------------------------------------
extern "C" void kernel_launch(void* const* d_in, const int* in_sizes, int n_in,
                              void* d_out, int out_size, void* d_ws, size_t ws_size,
                              hipStream_t stream) {
  const int*   old_tok = (const int*)  d_in[0];
  const float* emb     = (const float*)d_in[1];
  const float* W_ih    = (const float*)d_in[2];
  const float* W_hh    = (const float*)d_in[3];
  const float* b_ih    = (const float*)d_in[4];
  const float* b_hh    = (const float*)d_in[5];
  const float* W_lin   = (const float*)d_in[6];
  const float* b_lin   = (const float*)d_in[7];
  float* out = (float*)d_out;

  char* ws = (char*)d_ws;
  size_t off = 0;
  auto carve = [&](size_t bytes) -> char* {
    char* p = ws + off;
    off = (off + bytes + 255) & ~(size_t)255;
    return p;
  };
  const int  ih_elems  = 64   * 8 * 32 * 16;     // 262,144
  const int  lin_elems = 2000 * 8 * 32 * 16;     // 8,192,000
  const long gx_elems  = (long)TSTEPS * 16 * 1024;
  const long hs_elems  = (long)TSTEPS * 8 * 32 * 16 + 4096; // +1 M-tile slack
  __bf16* WihP  = (__bf16*)carve((size_t)ih_elems  * 2);
  __bf16* WhhP  = (__bf16*)carve((size_t)ih_elems  * 2);
  __bf16* WlinP = (__bf16*)carve((size_t)lin_elems * 2);
  float*  Gx    = (float*) carve((size_t)gx_elems  * 4);
  __bf16* hsP   = (__bf16*)carve((size_t)hs_elems  * 2);

  pack_b_bf16<<<(ih_elems  + 255) / 256, 256, 0, stream>>>(W_ih,  WihP,  ih_elems);
  pack_b_bf16<<<(ih_elems  + 255) / 256, 256, 0, stream>>>(W_hh,  WhhP,  ih_elems);
  pack_b_bf16<<<(lin_elems + 255) / 256, 256, 0, stream>>>(W_lin, WlinP, lin_elems);

  gx_kernel<<<(TSTEPS * 64) / 8, 256, 0, stream>>>(old_tok, emb, WihP, b_ih, b_hh, Gx);

  lstm_rec_kernel<<<1, 1024, 0, stream>>>(Gx, WhhP, hsP);

  dim3 lgrid(63, 5);
  logits_kernel<<<lgrid, 256, 0, stream>>>(hsP, WlinP, b_lin, out);
}
